// TGCNmodel_10591389352453
// MI455X (gfx1250) — matrix operand order repaired and checked by reading, not compiled, not run on hardware
//
#include <hip/hip_runtime.h>
#include <hip/hip_bf16.h>
#include <math.h>

typedef __attribute__((ext_vector_type(16))) _Float16 v16h;
typedef __attribute__((ext_vector_type(8)))  float    v8f;

#define BB 32
#define TT 256
#define NN 1000
#define EE 2000
#define HEADS 8
#define CC1 3
#define CC2 2
#define HID1 24
#define HID2 16
#define HLN 512
#define BN   (BB*NN)      /* 32000 */
#define EBAT (BB*EE)      /* 64000 */
#define EALL (EBAT+BN)    /* 96000 */
#define INFLOW 753
#define KPAD 2016
#define KT1 63            /* 2016/32 */
#define KT2 16            /* 512/32  */
#define NTG 128           /* 2048/16 */
#define NTO 125           /* 2000/16 */
#define PRED_SZ ((size_t)BB*TT*NN*2)   /* 16,384,000 */

__device__ __forceinline__ float lrelu(float x, float s){ return x > 0.f ? x : s*x; }
__device__ __forceinline__ float sigm(float x){ return 1.f/(1.f + __expf(-x)); }

__device__ __forceinline__ void atomicMaxF(float* addr, float val){
  unsigned int* ua = reinterpret_cast<unsigned int*>(addr);
  unsigned int cur = __float_as_uint(*addr);
  while (val > __uint_as_float(cur)){
    unsigned int prev = atomicCAS(ua, cur, __float_as_uint(val));
    if (prev == cur) break;
    cur = prev;
  }
}

__device__ __forceinline__ void edge_decode(int e, const int* __restrict__ ei, int& si, int& di){
  if (e < EBAT){
    int b = e / EE; int k = e - b*EE;
    si = b*NN + ei[k];
    di = b*NN + ei[EE + k];
  } else {
    si = di = e - EBAT;
  }
}

// A-fragment inverse map: element index for (m, k) in a (2 x KT) fragment array.
__device__ __forceinline__ int afrag_index(int m, int k, int KT){
  int kt = k >> 5, kl = k & 31, grp = kl >> 3;
  int lane = ((grp & 1) << 4) | (m & 15);
  int e = kl - 8*((grp + 1) >> 1);
  int mt = m >> 4;
  return ((mt*KT + kt)*32 + lane)*16 + e;
}

__global__ void k_zero(float* p, int n){
  int i = blockIdx.x*blockDim.x + threadIdx.x;
  if (i < n) p[i] = 0.f;
}

// Pack W (Nrows x Kcols, row-major, used as B[k,n] = W[n,k]) into WMMA B-fragments.
__global__ void k_pack_b(const float* __restrict__ W, _Float16* __restrict__ out,
                         int KT, int NT, int Kreal){
  int id = blockIdx.x*blockDim.x + threadIdx.x;
  if (id >= NT*KT*512) return;
  int e    = id & 15;
  int lane = (id >> 4) & 31;
  int kt   = (id >> 9) % KT;
  int nt   = (id >> 9) / KT;
  int n = nt*16 + (lane & 15);
  int k = kt*32 + ((lane >> 4) << 3) + (e & 7) + ((e >> 3) << 4);
  float v = (k < Kreal) ? W[(size_t)n*Kreal + k] : 0.f;
  out[id] = (_Float16)v;
}

// GAT layer 1 per-node: features, h1 = x@W1, attention logits, status; init scratch.
__global__ void k_gat1_node(const float* __restrict__ xn, const float* __restrict__ forcing, int t,
                            const float* __restrict__ W1, const float* __restrict__ as1,
                            const float* __restrict__ ad1,
                            float* __restrict__ h1, float* __restrict__ s1, float* __restrict__ d1,
                            int* __restrict__ status, float* __restrict__ mb,
                            float* __restrict__ den, float* __restrict__ agg){
  int i = blockIdx.x*blockDim.x + threadIdx.x;
  if (i >= BN) return;
  int b = i / NN, n = i - b*NN;
  float x0 = xn[i*2+0], x1 = xn[i*2+1];
  const float* f = forcing + (((size_t)b*TT + t)*NN + n)*2;
  float x2 = f[0], x3 = f[1];
  float hv[HID1];
  #pragma unroll
  for (int j = 0; j < HID1; ++j){
    hv[j] = x0*W1[j] + x1*W1[HID1 + j] + x2*W1[2*HID1 + j] + x3*W1[3*HID1 + j];
    h1[i*HID1 + j] = hv[j];
  }
  #pragma unroll
  for (int h = 0; h < HEADS; ++h){
    float s = 0.f, d = 0.f;
    #pragma unroll
    for (int c = 0; c < CC1; ++c){
      s += hv[h*CC1 + c] * as1[h*CC1 + c];
      d += hv[h*CC1 + c] * ad1[h*CC1 + c];
    }
    s1[i*HEADS + h] = s;
    d1[i*HEADS + h] = d;
    mb[i*HEADS + h] = -1e30f;
    den[i*HEADS + h] = 0.f;
  }
  int st = (x0 > 0.f) || (x1 > 0.f);
  if (n == INFLOW) st = st || (x3 > 0.f);
  status[i] = st;
  #pragma unroll
  for (int j = 0; j < HID1; ++j) agg[i*HID1 + j] = 0.f;
}

__global__ void k_edge_max(const int* __restrict__ ei, const int* __restrict__ status,
                           const float* __restrict__ s, const float* __restrict__ d,
                           float* __restrict__ mb){
  int e = blockIdx.x*blockDim.x + threadIdx.x;
  if (e >= EALL) return;
  int si, di; edge_decode(e, ei, si, di);
  int mask = (e < EBAT) ? status[si] : 1;
  float4 sa = *(const float4*)&s[si*HEADS];
  float4 sb = *(const float4*)&s[si*HEADS + 4];
  float4 da = *(const float4*)&d[di*HEADS];
  float4 db = *(const float4*)&d[di*HEADS + 4];
  float sv[HEADS] = {sa.x, sa.y, sa.z, sa.w, sb.x, sb.y, sb.z, sb.w};
  float dv[HEADS] = {da.x, da.y, da.z, da.w, db.x, db.y, db.z, db.w};
  #pragma unroll
  for (int h = 0; h < HEADS; ++h){
    float v = lrelu(sv[h] + dv[h], 0.2f);
    if (!mask) v = -1e30f;
    atomicMaxF(&mb[di*HEADS + h], v);
  }
}

__global__ void k_edge_sum(const int* __restrict__ ei, const int* __restrict__ status,
                           const float* __restrict__ s, const float* __restrict__ d,
                           const float* __restrict__ mb, float* __restrict__ den,
                           float* __restrict__ ee){
  int e = blockIdx.x*blockDim.x + threadIdx.x;
  if (e >= EALL) return;
  int si, di; edge_decode(e, ei, si, di);
  int mask = (e < EBAT) ? status[si] : 1;
  float4 sa = *(const float4*)&s[si*HEADS];
  float4 sb = *(const float4*)&s[si*HEADS + 4];
  float4 da = *(const float4*)&d[di*HEADS];
  float4 db = *(const float4*)&d[di*HEADS + 4];
  float4 ma = *(const float4*)&mb[di*HEADS];
  float4 mc = *(const float4*)&mb[di*HEADS + 4];
  float sv[HEADS] = {sa.x, sa.y, sa.z, sa.w, sb.x, sb.y, sb.z, sb.w};
  float dv[HEADS] = {da.x, da.y, da.z, da.w, db.x, db.y, db.z, db.w};
  float mv[HEADS] = {ma.x, ma.y, ma.z, ma.w, mc.x, mc.y, mc.z, mc.w};
  float val[HEADS];
  #pragma unroll
  for (int h = 0; h < HEADS; ++h){
    val[h] = 0.f;
    if (mask){
      float v = lrelu(sv[h] + dv[h], 0.2f);
      val[h] = __expf(v - mv[h]);
      atomicAdd(&den[di*HEADS + h], val[h]);
    }
  }
  *(float4*)&ee[(size_t)e*HEADS]     = make_float4(val[0], val[1], val[2], val[3]);
  *(float4*)&ee[(size_t)e*HEADS + 4] = make_float4(val[4], val[5], val[6], val[7]);
}

__global__ void k_edge_agg(const int* __restrict__ ei, const float* __restrict__ ee,
                           const float* __restrict__ den, const float* __restrict__ hsrc,
                           float* __restrict__ agg, int C, int hstride, int astride,
                           float* __restrict__ alphaOut){
  int e = blockIdx.x*blockDim.x + threadIdx.x;
  if (e >= EALL) return;
  int si, di; edge_decode(e, ei, si, di);
  float4 ea = *(const float4*)&ee[(size_t)e*HEADS];
  float4 eb = *(const float4*)&ee[(size_t)e*HEADS + 4];
  float4 na = *(const float4*)&den[di*HEADS];
  float4 nb = *(const float4*)&den[di*HEADS + 4];
  float ev[HEADS] = {ea.x, ea.y, ea.z, ea.w, eb.x, eb.y, eb.z, eb.w};
  float nv[HEADS] = {na.x, na.y, na.z, na.w, nb.x, nb.y, nb.z, nb.w};
  float a[HEADS];
  #pragma unroll
  for (int h = 0; h < HEADS; ++h) a[h] = ev[h] / (nv[h] + 1e-16f);
  if (alphaOut){
    *(float4*)&alphaOut[(size_t)e*HEADS]     = make_float4(a[0], a[1], a[2], a[3]);
    *(float4*)&alphaOut[(size_t)e*HEADS + 4] = make_float4(a[4], a[5], a[6], a[7]);
  }
  for (int h = 0; h < HEADS; ++h)
    for (int c = 0; c < C; ++c)
      atomicAdd(&agg[di*astride + h*C + c], a[h] * hsrc[si*hstride + h*C + c]);
}

// finalize GAT1 (bias + leaky), then per-node part of GAT2; re-init scratch.
__global__ void k_gat_mid(float* __restrict__ agg, const float* __restrict__ b1,
                          const float* __restrict__ W2, const float* __restrict__ as2,
                          const float* __restrict__ ad2,
                          float* __restrict__ h2, float* __restrict__ s2, float* __restrict__ d2,
                          float* __restrict__ mb, float* __restrict__ den){
  int i = blockIdx.x*blockDim.x + threadIdx.x;
  if (i >= BN) return;
  float g1v[HID1];
  #pragma unroll
  for (int j = 0; j < HID1; ++j) g1v[j] = lrelu(agg[i*HID1 + j] + b1[j], 0.01f);
  float hv[HID2];
  #pragma unroll
  for (int j2 = 0; j2 < HID2; ++j2){
    float s = 0.f;
    #pragma unroll
    for (int j = 0; j < HID1; ++j) s += g1v[j] * W2[j*HID2 + j2];
    hv[j2] = s;
    h2[i*HID2 + j2] = s;
  }
  #pragma unroll
  for (int h = 0; h < HEADS; ++h){
    float s = 0.f, d = 0.f;
    #pragma unroll
    for (int c = 0; c < CC2; ++c){
      s += hv[h*CC2 + c] * as2[h*CC2 + c];
      d += hv[h*CC2 + c] * ad2[h*CC2 + c];
    }
    s2[i*HEADS + h] = s;
    d2[i*HEADS + h] = d;
    mb[i*HEADS + h] = -1e30f;
    den[i*HEADS + h] = 0.f;
  }
  #pragma unroll
  for (int j = 0; j < HID1; ++j) agg[i*HID1 + j] = 0.f;
}

// finalize GAT2: mean over heads, bias, leaky -> write straight into A-fragment xfA.
__global__ void k_gat2_fin(const float* __restrict__ agg, const float* __restrict__ b2,
                           _Float16* __restrict__ xfA){
  int i = blockIdx.x*blockDim.x + threadIdx.x;
  if (i >= BN) return;
  int b = i / NN, n = i - b*NN;
  #pragma unroll
  for (int c = 0; c < CC2; ++c){
    float s = 0.f;
    #pragma unroll
    for (int h = 0; h < HEADS; ++h) s += agg[i*HID2 + h*CC2 + c];
    float g = lrelu(s * (1.f/HEADS) + b2[c], 0.01f);
    xfA[afrag_index(b, n*2 + c, KT1)] = (_Float16)g;
  }
}

// gates(32x2048) = xf @ W_ih^T + h @ W_hh^T + biases   (WMMA f16 -> f32)
__global__ void k_gemm_gates(const _Float16* __restrict__ xfA, const _Float16* __restrict__ WihP,
                             const _Float16* __restrict__ hnA, const _Float16* __restrict__ WhhP,
                             const float* __restrict__ b_ih, const float* __restrict__ b_hh,
                             float* __restrict__ gates){
  int lane = threadIdx.x;
  int nt = blockIdx.x, mt = blockIdx.y;
  v8f acc = {};
  const v16h* A  = (const v16h*)xfA  + (size_t)(mt*KT1)*32 + lane;
  const v16h* Bw = (const v16h*)WihP + (size_t)(nt*KT1)*32 + lane;
  for (int kt = 0; kt < KT1; ++kt){
    if (kt + 4 < KT1) __builtin_prefetch((const char*)(Bw + (kt+4)*32), 0, 1);
    acc = __builtin_amdgcn_wmma_f32_16x16x32_f16(false, A[kt*32], false, Bw[kt*32],
                                                 (short)0, acc, false, false);
  }
  const v16h* A2 = (const v16h*)hnA  + (size_t)(mt*KT2)*32 + lane;
  const v16h* B2 = (const v16h*)WhhP + (size_t)(nt*KT2)*32 + lane;
  for (int kt = 0; kt < KT2; ++kt){
    acc = __builtin_amdgcn_wmma_f32_16x16x32_f16(false, A2[kt*32], false, B2[kt*32],
                                                 (short)0, acc, false, false);
  }
  int col = nt*16 + (lane & 15);
  float bias = b_ih[col] + b_hh[col];
  int r0 = mt*16 + ((lane >> 4) << 3);
  #pragma unroll
  for (int r = 0; r < 8; ++r)
    gates[(r0 + r)*(4*HLN) + col] = acc[r] + bias;
}

// LSTM cell + LayerNorm; packs h directly into A-fragment layout.
__global__ void k_cell(const float* __restrict__ gates, float* __restrict__ cn,
                       const float* __restrict__ ln_g, const float* __restrict__ ln_b,
                       _Float16* __restrict__ hnA){
  int b = blockIdx.x;
  int j = threadIdx.x;
  const float* g = gates + (size_t)b*(4*HLN);
  float i_ = g[j], f_ = g[HLN + j], gg = g[2*HLN + j], o_ = g[3*HLN + j];
  float c = cn[b*HLN + j];
  c = sigm(f_)*c + sigm(i_)*tanhf(gg);
  cn[b*HLN + j] = c;
  float hraw = sigm(o_)*tanhf(c);
  __shared__ float red[HLN];
  red[j] = hraw; __syncthreads();
  for (int s = HLN/2; s > 0; s >>= 1){ if (j < s) red[j] += red[j + s]; __syncthreads(); }
  float mu = red[0] * (1.f/HLN);
  __syncthreads();
  float dv = hraw - mu;
  red[j] = dv*dv; __syncthreads();
  for (int s = HLN/2; s > 0; s >>= 1){ if (j < s) red[j] += red[j + s]; __syncthreads(); }
  float var = red[0] * (1.f/HLN);
  float hn = dv * rsqrtf(var + 1e-5f) * ln_g[j] + ln_b[j];
  hnA[afrag_index(b, j, KT2)] = (_Float16)hn;
}

// pred(32x2000) = softplus(h @ lin_W^T + lin_b) -> output slice + next xn
__global__ void k_gemm_out(const _Float16* __restrict__ hnA, const _Float16* __restrict__ linP,
                           const float* __restrict__ lin_b, float* __restrict__ pred,
                           float* __restrict__ xn, int t){
  int lane = threadIdx.x;
  int nt = blockIdx.x, mt = blockIdx.y;
  v8f acc = {};
  const v16h* A  = (const v16h*)hnA  + (size_t)(mt*KT2)*32 + lane;
  const v16h* Bw = (const v16h*)linP + (size_t)(nt*KT2)*32 + lane;
  for (int kt = 0; kt < KT2; ++kt){
    acc = __builtin_amdgcn_wmma_f32_16x16x32_f16(false, A[kt*32], false, Bw[kt*32],
                                                 (short)0, acc, false, false);
  }
  int col = nt*16 + (lane & 15);
  float bias = lin_b[col];
  int r0 = mt*16 + ((lane >> 4) << 3);
  #pragma unroll
  for (int r = 0; r < 8; ++r){
    int b = r0 + r;
    float v = acc[r] + bias;
    float sp = (v > 20.f) ? v : log1pf(__expf(v));
    pred[((size_t)b*TT + t)*(NN*2) + col] = sp;
    xn[b*(NN*2) + col] = sp;
  }
}

extern "C" void kernel_launch(void* const* d_in, const int* in_sizes, int n_in,
                              void* d_out, int out_size, void* d_ws, size_t ws_size,
                              hipStream_t stream) {
  (void)in_sizes; (void)n_in; (void)out_size; (void)ws_size;
  const float* forcing = (const float*)d_in[0];
  const int*   ei      = (const int*)d_in[1];
  const float* W1   = (const float*)d_in[2];
  const float* as1  = (const float*)d_in[3];
  const float* ad1  = (const float*)d_in[4];
  const float* b1   = (const float*)d_in[5];
  const float* W2   = (const float*)d_in[6];
  const float* as2  = (const float*)d_in[7];
  const float* ad2  = (const float*)d_in[8];
  const float* b2   = (const float*)d_in[9];
  const float* W_ih = (const float*)d_in[10];
  const float* W_hh = (const float*)d_in[11];
  const float* b_ih = (const float*)d_in[12];
  const float* b_hh = (const float*)d_in[13];
  const float* ln_g = (const float*)d_in[14];
  const float* ln_b = (const float*)d_in[15];
  const float* linW = (const float*)d_in[16];
  const float* linb = (const float*)d_in[17];
  float* outp = (float*)d_out;
  float* attn = outp + PRED_SZ;

  char* w = (char*)d_ws;
  size_t off = 0;
  auto take = [&](size_t bytes) -> char* {
    char* p = w + off;
    off = (off + bytes + 255) & ~(size_t)255;
    return p;
  };
  float* xn    = (float*)take((size_t)BN*2*4);
  float* cn    = (float*)take((size_t)BB*HLN*4);
  float* gates = (float*)take((size_t)BB*4*HLN*4);
  int*   status= (int*)  take((size_t)BN*4);
  float* h1    = (float*)take((size_t)BN*HID1*4);
  float* s1    = (float*)take((size_t)BN*HEADS*4);
  float* d1    = (float*)take((size_t)BN*HEADS*4);
  float* mb    = (float*)take((size_t)BN*HEADS*4);
  float* den   = (float*)take((size_t)BN*HEADS*4);
  float* agg   = (float*)take((size_t)BN*HID1*4);
  float* eebuf = (float*)take((size_t)EALL*HEADS*4);
  _Float16* xfA  = (_Float16*)take((size_t)2*KT1*512*2);
  _Float16* hnA  = (_Float16*)take((size_t)2*KT2*512*2);
  _Float16* WihP = (_Float16*)take((size_t)NTG*KT1*512*2);
  _Float16* WhhP = (_Float16*)take((size_t)NTG*KT2*512*2);
  _Float16* linP = (_Float16*)take((size_t)NTO*KT2*512*2);

  // one-time init + weight packing
  k_zero<<<(BN*2 + 255)/256, 256, 0, stream>>>(xn, BN*2);
  k_zero<<<(BB*HLN + 255)/256, 256, 0, stream>>>(cn, BB*HLN);
  k_zero<<<(2*KT1*512/2 + 255)/256, 256, 0, stream>>>((float*)xfA, 2*KT1*512/2);
  k_zero<<<(2*KT2*512/2 + 255)/256, 256, 0, stream>>>((float*)hnA, 2*KT2*512/2);
  k_pack_b<<<(NTG*KT1*512 + 255)/256, 256, 0, stream>>>(W_ih, WihP, KT1, NTG, NN*2);
  k_pack_b<<<(NTG*KT2*512 + 255)/256, 256, 0, stream>>>(W_hh, WhhP, KT2, NTG, HLN);
  k_pack_b<<<(NTO*KT2*512 + 255)/256, 256, 0, stream>>>(linW, linP, KT2, NTO, HLN);

  const int nbN = (BN + 255)/256;
  const int nbE = (EALL + 255)/256;
  for (int t = 0; t < TT; ++t){
    k_gat1_node<<<nbN, 256, 0, stream>>>(xn, forcing, t, W1, as1, ad1,
                                         h1, s1, d1, status, mb, den, agg);
    k_edge_max<<<nbE, 256, 0, stream>>>(ei, status, s1, d1, mb);
    k_edge_sum<<<nbE, 256, 0, stream>>>(ei, status, s1, d1, mb, den, eebuf);
    k_edge_agg<<<nbE, 256, 0, stream>>>(ei, eebuf, den, h1, agg, CC1, HID1, HID1,
                                        attn + (size_t)t*EALL*HEADS);
    k_gat_mid<<<nbN, 256, 0, stream>>>(agg, b1, W2, as2, ad2, h1, s1, d1, mb, den);
    k_edge_max<<<nbE, 256, 0, stream>>>(ei, status, s1, d1, mb);
    k_edge_sum<<<nbE, 256, 0, stream>>>(ei, status, s1, d1, mb, den, eebuf);
    k_edge_agg<<<nbE, 256, 0, stream>>>(ei, eebuf, den, h1, agg, CC2, HID2, HID2, nullptr);
    k_gat2_fin<<<nbN, 256, 0, stream>>>(agg, b2, xfA);
    dim3 gg(NTG, 2);
    k_gemm_gates<<<gg, 32, 0, stream>>>(xfA, WihP, hnA, WhhP, b_ih, b_hh, gates);
    k_cell<<<BB, HLN, 0, stream>>>(gates, cn, ln_g, ln_b, hnA);
    dim3 go(NTO, 2);
    k_gemm_out<<<go, 32, 0, stream>>>(hnA, linP, linb, outp, xn, t);
  }
}
